// FBPINNWithWindow_30940944401055
// MI455X (gfx1250) — matrix operand order repaired and checked by reading, not compiled, not run on hardware
//
#include <hip/hip_runtime.h>
#include <math.h>

typedef __attribute__((ext_vector_type(16))) _Float16 v16h;
typedef __attribute__((ext_vector_type(8)))  _Float16 v8h;
typedef __attribute__((ext_vector_type(8)))  float    v8f;

#define P_PTS 65536
#define J_SUB 16
#define WID   64
#define ASTRIDE 72            // padded act row stride (halves): conflict-free ds_load_b128
#define WTSTRIDE 17           // padded softmax tile stride (floats)
#define WAVES_PER_BLOCK 4
#define PTS_PER_WAVE    32

// d_ws layout (in _Float16 units): f16 B-fragments for every 64-wide GEMM.
// One 64x64 matrix -> 8 fragments (kc=0..1, nt=0..3); each fragment = 32 lanes x 16 halves.
#define POU_WH_OFF 0                    // 4 matrices  * 4096
#define POU_WL_OFF (4*4096)             // 1 (64x16)   * 1024
#define SUB_WH_OFF (4*4096 + 1024)      // 32 matrices * 4096
#define TOTAL_FRAG_HALVES (4*4096 + 1024 + 32*4096)

// ---------------------------------------------------------------------------
// Prep: convert f32 weights -> f16 fragments for WMMA B; k order below is this code's choice (A matches).
// B 32x16 (f16): lanes 0-15 hold K=kc*32+0..15 (halves ascending), lanes 16-31 hold K=kc*32+16..31.
// ---------------------------------------------------------------------------
__global__ __launch_bounds__(256) void prep_frags(
    const float* __restrict__ pouWh,   // [4][64][64]
    const float* __restrict__ pouWl,   // [64][16]
    const float* __restrict__ subWh,   // [16][2][64][64]
    _Float16* __restrict__ frags)
{
  int t = blockIdx.x * blockDim.x + threadIdx.x;
  if (t >= TOTAL_FRAG_HALVES) return;
  float val;
  if (t < POU_WL_OFF) {                               // pou_Wh fragments
    int mat = t >> 12, r = t & 4095;
    int f = r >> 9, e = r & 511;
    int kc = f >> 2, nt = f & 3;
    int lane = e >> 4, i = e & 15;
    int k = kc * 32 + ((lane >> 4) << 4) + i;
    int n = nt * 16 + (lane & 15);
    val = pouWh[mat * 4096 + k * 64 + n];
  } else if (t < SUB_WH_OFF) {                        // pou_Wl fragments (64x16)
    int r = t - POU_WL_OFF;
    int kc = r >> 9, e = r & 511;
    int lane = e >> 4, i = e & 15;
    int k = kc * 32 + ((lane >> 4) << 4) + i;
    int n = lane & 15;
    val = pouWl[k * 16 + n];
  } else {                                            // sub_Wh fragments
    int r = t - SUB_WH_OFF;
    int mat = r >> 12; r &= 4095;
    int f = r >> 9, e = r & 511;
    int kc = f >> 2, nt = f & 3;
    int lane = e >> 4, i = e & 15;
    int k = kc * 32 + ((lane >> 4) << 4) + i;
    int n = nt * 16 + (lane & 15);
    val = subWh[mat * 4096 + k * 64 + n];
  }
  frags[t] = (_Float16)val;
}

// ---------------------------------------------------------------------------
// Helpers
// ---------------------------------------------------------------------------
__device__ __forceinline__ v8f wmma_f16(v16h a, v16h b, v8f c) {
  return __builtin_amdgcn_wmma_f32_16x16x32_f16(false, a, false, b, (short)0, c, false, false);
}

// fast tanh, 5 VALU ops: 1 - 2/(exp2(2*log2e*x)+1); exact +-1 saturation
__device__ __forceinline__ float fast_tanh(float v) {
  float e = __builtin_amdgcn_exp2f(v * 2.8853900817779268f);
  return fmaf(-2.0f, __builtin_amdgcn_rcpf(e + 1.0f), 1.0f);
}

// A 16x32 f16 from LDS act tile: lane<16 -> K = kc*32 + 0..15; lane>=16 -> +16
__device__ __forceinline__ v16h lds_afrag(const _Float16* act, int mt, int kc, int lane) {
  int m = mt * 16 + (lane & 15);
  int b = m * ASTRIDE + kc * 32 + ((lane >> 4) << 4);
  union { v16h v; v8h q[2]; } u;
  u.q[0] = *(const v8h*)(act + b);        // 16B, 16B-aligned
  u.q[1] = *(const v8h*)(act + b + 8);
  return u.v;
}

// B fragment: 32 contiguous bytes per lane, prepped by prep_frags
__device__ __forceinline__ v16h g_bfrag(const _Float16* fragbase, int f, int lane) {
  return *((const v16h*)fragbase + f * 32 + lane);
}

struct WaveScratch {
  _Float16 act[PTS_PER_WAVE * ASTRIDE];   // activation tile (A-source), f16, padded
  float    wt[PTS_PER_WAVE * WTSTRIDE];   // POU softmax weights [point][j], padded
  float    xs[PTS_PER_WAVE * 2];          // (x-0.5)*2
  float    xr[PTS_PER_WAVE * 2];          // raw x
};

// ---------------------------------------------------------------------------
// Main kernel: one wave = 32 points (two M-tiles), full pipeline per wave.
// Waves are fully independent (same-wave DS ops are in-order on CDNA5).
// ---------------------------------------------------------------------------
__global__ __launch_bounds__(WAVES_PER_BLOCK * 32) void fbpinn_kernel(
    const float* __restrict__ x,
    const float* __restrict__ sub_W0, const float* __restrict__ sub_b0,
    const float* __restrict__ sub_bh,
    const float* __restrict__ sub_Wl, const float* __restrict__ sub_bl,
    const float* __restrict__ pou_W0, const float* __restrict__ pou_b0,
    const float* __restrict__ pou_bh, const float* __restrict__ pou_bl,
    const _Float16* __restrict__ frags,
    float* __restrict__ out)
{
  __shared__ WaveScratch sc[WAVES_PER_BLOCK];
  const int lane  = threadIdx.x & 31;
  const int half8 = (lane >> 4) << 3;     // 0 or 8 (C-layout M offset)
  const int n0    = lane & 15;            // C/B-layout N index
  const int wv    = threadIdx.x >> 5;
  const int base  = (blockIdx.x * WAVES_PER_BLOCK + wv) * PTS_PER_WAVE;
  WaveScratch& S = sc[wv];

  // warm L2 with the fragment pool (gfx1250 global_prefetch_b8)
  __builtin_prefetch(frags + SUB_WH_OFF + lane * 64, 0, 1);

  {
    float x0 = x[(base + lane) * 2 + 0];
    float x1 = x[(base + lane) * 2 + 1];
    S.xr[lane * 2 + 0] = x0;
    S.xr[lane * 2 + 1] = x1;
    S.xs[lane * 2 + 0] = (x0 - 0.5f) * 2.0f;
    S.xs[lane * 2 + 1] = (x1 - 0.5f) * 2.0f;
  }

  // Per-lane copies of the coordinates of the 16 C-layout points this lane owns
  // (m = mt*16 + r + half8), fetched as 16 ds_load_b64.
  float xr0[2][8], xr1[2][8];
#pragma unroll
  for (int mt = 0; mt < 2; ++mt)
#pragma unroll
    for (int r = 0; r < 8; ++r) {
      int m = mt * 16 + r + half8;
      xr0[mt][r] = S.xr[m * 2 + 0];
      xr1[mt][r] = S.xr[m * 2 + 1];
    }

  // ---- POU input layer, computed directly in C layout: h = relu(x @ W0 + b0) ----
  v8f hreg[2][4];                          // f32 residual trunk in registers
#pragma unroll
  for (int nt = 0; nt < 4; ++nt) {
    float wa = pou_W0[nt * 16 + n0];
    float wb = pou_W0[64 + nt * 16 + n0];
    float bb = pou_b0[nt * 16 + n0];
#pragma unroll
    for (int mt = 0; mt < 2; ++mt)
#pragma unroll
      for (int r = 0; r < 8; ++r) {
        int m = mt * 16 + r + half8;
        float h = fmaxf(fmaf(xr0[mt][r], wa, fmaf(xr1[mt][r], wb, bb)), 0.0f);
        hreg[mt][nt][r] = h;
        S.act[m * ASTRIDE + nt * 16 + n0] = (_Float16)h;
      }
  }

  // ---- POU residual blocks: h = h + relu(h @ Wh[i] + bh[i]) ----
  for (int blk = 0; blk < 4; ++blk) {
    v16h a00 = lds_afrag(S.act, 0, 0, lane), a01 = lds_afrag(S.act, 0, 1, lane);
    v16h a10 = lds_afrag(S.act, 1, 0, lane), a11 = lds_afrag(S.act, 1, 1, lane);
    const _Float16* fb = frags + POU_WH_OFF + blk * 4096;
#pragma unroll
    for (int nt = 0; nt < 4; ++nt) {
      v16h b0 = g_bfrag(fb, nt, lane), b1 = g_bfrag(fb, 4 + nt, lane);
      float bv = pou_bh[blk * 64 + nt * 16 + n0];
#pragma unroll
      for (int mt = 0; mt < 2; ++mt) {
        v8f g = {bv, bv, bv, bv, bv, bv, bv, bv};
        g = wmma_f16(mt ? a10 : a00, b0, g);
        g = wmma_f16(mt ? a11 : a01, b1, g);
#pragma unroll
        for (int r = 0; r < 8; ++r) {
          float hv = hreg[mt][nt][r] + fmaxf(g[r], 0.0f);
          hreg[mt][nt][r] = hv;
          S.act[(mt * 16 + r + half8) * ASTRIDE + nt * 16 + n0] = (_Float16)hv;
        }
      }
    }
  }

  // ---- POU logits: h @ Wl + bl ----
  {
    v16h a00 = lds_afrag(S.act, 0, 0, lane), a01 = lds_afrag(S.act, 0, 1, lane);
    v16h a10 = lds_afrag(S.act, 1, 0, lane), a11 = lds_afrag(S.act, 1, 1, lane);
    const _Float16* fb = frags + POU_WL_OFF;
    v16h b0 = g_bfrag(fb, 0, lane), b1 = g_bfrag(fb, 1, lane);
    float bv = pou_bl[n0];
#pragma unroll
    for (int mt = 0; mt < 2; ++mt) {
      v8f c = {bv, bv, bv, bv, bv, bv, bv, bv};
      c = wmma_f16(mt ? a10 : a00, b0, c);
      c = wmma_f16(mt ? a11 : a01, b1, c);
#pragma unroll
      for (int r = 0; r < 8; ++r)
        S.wt[(mt * 16 + r + half8) * WTSTRIDE + n0] = c[r];
    }
  }

  // ---- softmax over J per point (point = lane, all 32 lanes busy) ----
  {
    float mx = -3.0e38f;
#pragma unroll
    for (int j = 0; j < J_SUB; ++j) mx = fmaxf(mx, S.wt[lane * WTSTRIDE + j]);
    float s = 0.0f;
#pragma unroll
    for (int j = 0; j < J_SUB; ++j) {
      float e = __expf(S.wt[lane * WTSTRIDE + j] - mx);
      S.wt[lane * WTSTRIDE + j] = e;
      s += e;
    }
    float inv = 1.0f / s;
#pragma unroll
    for (int j = 0; j < J_SUB; ++j) S.wt[lane * WTSTRIDE + j] *= inv;
  }

  // Per-lane copies of scaled coordinates for the subnet input layers
  float xs0[2][8], xs1[2][8];
#pragma unroll
  for (int mt = 0; mt < 2; ++mt)
#pragma unroll
    for (int r = 0; r < 8; ++r) {
      int m = mt * 16 + r + half8;
      xs0[mt][r] = S.xs[m * 2 + 0];
      xs1[mt][r] = S.xs[m * 2 + 1];
    }

  float total = 0.0f;   // point = lane

  // ---- subnets: tanh MLP per subdomain, accumulate w[p][j] * u_j[p] ----
  for (int j = 0; j < J_SUB; ++j) {
    const float* w0  = sub_W0 + j * 128;
    const float* b0p = sub_b0 + j * 64;
    // input layer 2 -> 64, tanh, computed directly in C layout
#pragma unroll
    for (int nt = 0; nt < 4; ++nt) {
      float wa = w0[nt * 16 + n0];
      float wb = w0[64 + nt * 16 + n0];
      float bb = b0p[nt * 16 + n0];
#pragma unroll
      for (int mt = 0; mt < 2; ++mt)
#pragma unroll
        for (int r = 0; r < 8; ++r) {
          int m = mt * 16 + r + half8;
          float h = fmaf(xs0[mt][r], wa, fmaf(xs1[mt][r], wb, bb));
          S.act[m * ASTRIDE + nt * 16 + n0] = (_Float16)fast_tanh(h);
        }
    }
    // two hidden layers 64 -> 64 (WMMA), tanh
#pragma unroll
    for (int L = 0; L < 2; ++L) {
      v16h a00 = lds_afrag(S.act, 0, 0, lane), a01 = lds_afrag(S.act, 0, 1, lane);
      v16h a10 = lds_afrag(S.act, 1, 0, lane), a11 = lds_afrag(S.act, 1, 1, lane);
      const _Float16* fb = frags + SUB_WH_OFF + (j * 2 + L) * 4096;
      const float* bh = sub_bh + (j * 2 + L) * 64;
#pragma unroll
      for (int nt = 0; nt < 4; ++nt) {
        v16h b0 = g_bfrag(fb, nt, lane), b1 = g_bfrag(fb, 4 + nt, lane);
        float bv = bh[nt * 16 + n0];
#pragma unroll
        for (int mt = 0; mt < 2; ++mt) {
          v8f c = {bv, bv, bv, bv, bv, bv, bv, bv};
          c = wmma_f16(mt ? a10 : a00, b0, c);
          c = wmma_f16(mt ? a11 : a01, b1, c);
#pragma unroll
          for (int r = 0; r < 8; ++r)
            S.act[(mt * 16 + r + half8) * ASTRIDE + nt * 16 + n0] = (_Float16)fast_tanh(c[r]);
        }
      }
    }
    // output layer 64 -> 1 + weighted accumulate (point = lane, all 32 lanes)
    {
      float u = sub_bl[j];
      const float* wl = sub_Wl + j * 64;   // uniform across lanes -> SMEM loads
      union { v8h v; _Float16 h[8]; } ua;
#pragma unroll
      for (int c8 = 0; c8 < 8; ++c8) {
        ua.v = *(const v8h*)&S.act[lane * ASTRIDE + c8 * 8];   // 16B, 16B-aligned
#pragma unroll
        for (int w = 0; w < 8; ++w)
          u = fmaf((float)ua.h[w], wl[c8 * 8 + w], u);
      }
      total = fmaf(S.wt[lane * WTSTRIDE + j], u, total);
    }
  }

  // ---- ansatz: hard zero-Dirichlet BC ----
  {
    float x0 = S.xr[lane * 2], x1 = S.xr[lane * 2 + 1];
    const float pi = 3.14159265358979323846f;
    out[base + lane] = total * sinf(pi * x0) * sinf(pi * x1);
  }
}

// ---------------------------------------------------------------------------
extern "C" void kernel_launch(void* const* d_in, const int* in_sizes, int n_in,
                              void* d_out, int out_size, void* d_ws, size_t ws_size,
                              hipStream_t stream) {
  (void)in_sizes; (void)n_in; (void)out_size; (void)ws_size;
  const float* x      = (const float*)d_in[0];
  const float* sub_W0 = (const float*)d_in[1];
  const float* sub_b0 = (const float*)d_in[2];
  const float* sub_Wh = (const float*)d_in[3];
  const float* sub_bh = (const float*)d_in[4];
  const float* sub_Wl = (const float*)d_in[5];
  const float* sub_bl = (const float*)d_in[6];
  const float* pou_W0 = (const float*)d_in[7];
  const float* pou_b0 = (const float*)d_in[8];
  const float* pou_Wh = (const float*)d_in[9];
  const float* pou_bh = (const float*)d_in[10];
  const float* pou_Wl = (const float*)d_in[11];
  const float* pou_bl = (const float*)d_in[12];
  _Float16* frags = (_Float16*)d_ws;

  prep_frags<<<(TOTAL_FRAG_HALVES + 255) / 256, 256, 0, stream>>>(pou_Wh, pou_Wl, sub_Wh, frags);

  int blocks = P_PTS / (WAVES_PER_BLOCK * PTS_PER_WAVE);   // 512
  fbpinn_kernel<<<blocks, WAVES_PER_BLOCK * 32, 0, stream>>>(
      x, sub_W0, sub_b0, sub_bh, sub_Wl, sub_bl,
      pou_W0, pou_b0, pou_bh, pou_bl, frags, (float*)d_out);
}
